// MFA_33569464385914
// MI455X (gfx1250) — matrix operand order repaired and checked by reading, compiled, not run on hardware
//
#include <hip/hip_runtime.h>
#include <hip/hip_bf16.h>
#include <math.h>

// ---------------- types & constants ----------------
typedef float  v8f   __attribute__((ext_vector_type(8)));
typedef __bf16 v16bf __attribute__((ext_vector_type(16)));

#define KK 512
#define DD 2048
#define QQ 8
#define BB 8192
#define KSPLIT 8              // K loop split across blockIdx.y
#define EPSF 1e-5f
#define LOG2PI 1.8378770664093453f

// ---------------- helpers ----------------
__device__ __forceinline__ unsigned short f2bf(float f) {
  unsigned int u = __float_as_uint(f);
  u += 0x7FFFu + ((u >> 16) & 1u);       // round-to-nearest-even
  return (unsigned short)(u >> 16);
}

__device__ __forceinline__ float softplusf(float x) {
  return (x > 20.0f) ? x : log1pf(__expf(x));
}

__device__ __forceinline__ float block_reduce256(float v, float* red) {
  const int t = threadIdx.x;
  red[t] = v; __syncthreads();
  #pragma unroll
  for (int off = 128; off > 0; off >>= 1) {
    if (t < off) red[t] += red[t + off];
    __syncthreads();
  }
  float r = red[0];
  __syncthreads();
  return r;
}

// Async fill of one 9216B stage buffer (8 PinvW B-frags + 1 mu B-frag)
// using CDNA5 async global->LDS loads (ASYNCcnt tracked, no VGPR round trip).
__device__ __forceinline__ void async_stage(unsigned short* dstLds,
                                            const unsigned short* srcW,
                                            const unsigned short* srcM,
                                            int t) {
  unsigned lds0 = (unsigned)(uintptr_t)dstLds + (unsigned)t * 16u;
  const char* g0 = (const char*)srcW + (size_t)t * 16;
  asm volatile("global_load_async_to_lds_b128 %0, %1, off"
               :: "v"(lds0), "v"(g0) : "memory");
  unsigned lds1 = lds0 + 4096u;
  const char* g1 = (const char*)srcW + 4096 + (size_t)t * 16;
  asm volatile("global_load_async_to_lds_b128 %0, %1, off"
               :: "v"(lds1), "v"(g1) : "memory");
  if (t < 64) {
    unsigned lds2 = (unsigned)(uintptr_t)dstLds + 8192u + (unsigned)t * 16u;
    const char* g2 = (const char*)srcM + (size_t)t * 16;
    asm volatile("global_load_async_to_lds_b128 %0, %1, off"
                 :: "v"(lds2), "v"(g2) : "memory");
  }
}

// ---------------- kernel 1: psi prep ----------------
__global__ __launch_bounds__(256) void psi_kernel(const float* __restrict__ psi_rho,
                                                  float* __restrict__ psi_inv,
                                                  float* __restrict__ logdetp) {
  __shared__ float red[256];
  const int t = threadIdx.x;
  float acc = 0.0f;
  for (int d = t; d < DD; d += 256) {
    float sp = softplusf(psi_rho[d]) + EPSF;
    psi_inv[d] = 1.0f / sp;
    acc += logf(sp);
  }
  float s = block_reduce256(acc, red);
  if (t == 0) logdetp[0] = s;
}

// ---------------- kernel 2: pack x -> bf16 WMMA A-fragment order; xq = x^2 . psi_inv ----
__global__ __launch_bounds__(256) void pack_x_kernel(const float* __restrict__ x,
                                                     const float* __restrict__ psi_inv,
                                                     unsigned short* __restrict__ xp,
                                                     float* __restrict__ xq) {
  __shared__ float red[256];
  const int b = blockIdx.x;
  const int t = threadIdx.x;
  const float* xr = x + (size_t)b * DD;
  unsigned short* xpr = xp + ((size_t)(b >> 4)) * 64 * 512;
  const int lanelo = b & 15;
  float acc = 0.0f;
  #pragma unroll
  for (int it = 0; it < 8; ++it) {
    int d = t + it * 256;
    float xv = xr[d];
    acc += xv * xv * psi_inv[d];
    int dblk = d >> 5, dd = d & 31;
    int hi = dd >> 4, w16 = dd & 15, laneHi = w16 >> 3, ii = w16 & 7;
    xpr[(size_t)dblk * 512 + (size_t)(hi * 16 + lanelo) * 16 + w16] = f2bf(xv);
  }
  float s = block_reduce256(acc, red);
  if (t == 0) xq[b] = s;
}

// ---------------- kernel 3: per-k weight prep & packing ----------------
__global__ __launch_bounds__(256) void prep_w_kernel(const float* __restrict__ mu,
                                                     const float* __restrict__ dir_raw,
                                                     const float* __restrict__ scale_rho,
                                                     const float* __restrict__ psi_inv,
                                                     unsigned short* __restrict__ gwp,
                                                     unsigned short* __restrict__ gmp,
                                                     float* __restrict__ mut,   // K*36
                                                     float* __restrict__ wtpm,  // K*8
                                                     float* __restrict__ mupm)  // K
{
  __shared__ float red[256];
  __shared__ float coef[QQ];
  const int k = blockIdx.x;
  const int t = threadIdx.x;
  const float* dr  = dir_raw + (size_t)k * DD * QQ;
  const float* mur = mu + (size_t)k * DD;

  float ssq[QQ];
  #pragma unroll
  for (int q = 0; q < QQ; ++q) ssq[q] = 0.0f;
  float mq = 0.0f;
  for (int d = t; d < DD; d += 256) {
    float mv = mur[d];
    mq += mv * mv * psi_inv[d];
    #pragma unroll
    for (int q = 0; q < QQ; ++q) {
      float v = dr[(size_t)d * QQ + q];
      ssq[q] += v * v;
    }
  }
  #pragma unroll
  for (int q = 0; q < QQ; ++q) {
    float n2 = block_reduce256(ssq[q], red);
    if (t == 0) {
      float n = fmaxf(sqrtf(n2), EPSF);
      coef[q] = softplusf(scale_rho[k * QQ + q]) / n;
    }
  }
  float mqs = block_reduce256(mq, red);
  if (t == 0) mupm[k] = mqs;
  __syncthreads();

  float macc[36];
  #pragma unroll
  for (int e = 0; e < 36; ++e) macc[e] = 0.0f;
  float wacc[QQ];
  #pragma unroll
  for (int q = 0; q < QQ; ++q) wacc[q] = 0.0f;

  const int ct = k >> 1;
  const int nb = (k & 1) * 8;
  for (int d = t; d < DD; d += 256) {
    float pid = psi_inv[d];
    float mv  = mur[d];
    float wv[QQ], pw[QQ];
    #pragma unroll
    for (int q = 0; q < QQ; ++q) {
      wv[q] = dr[(size_t)d * QQ + q] * coef[q];
      pw[q] = pid * wv[q];
      wacc[q] += mv * pw[q];
    }
    int idx = 0;
    #pragma unroll
    for (int i = 0; i < QQ; ++i)
      #pragma unroll
      for (int j = 0; j < QQ; ++j)
        if (j >= i) macc[idx++] += wv[i] * wv[j] * pid;

    int dblk = d >> 5, dd = d & 31, laneHi = dd >> 4, i16 = dd & 15;
    size_t baseW = ((size_t)dblk * 256 + ct) * 512;
    #pragma unroll
    for (int q = 0; q < QQ; ++q)
      gwp[baseW + (size_t)(laneHi * 16 + nb + q) * 16 + i16] = f2bf(pw[q]);
    gmp[((size_t)dblk * 32 + (k >> 4)) * 512 + (size_t)(laneHi * 16 + (k & 15)) * 16 + i16]
        = f2bf(pid * mv);
  }
  #pragma unroll
  for (int e = 0; e < 36; ++e) {
    float s = block_reduce256(macc[e], red);
    if (t == 0) mut[(size_t)k * 36 + e] = s;
  }
  #pragma unroll
  for (int q = 0; q < QQ; ++q) {
    float s = block_reduce256(wacc[q], red);
    if (t == 0) wtpm[(size_t)k * QQ + q] = s;
  }
}

// ---------------- kernel 4: 8x8 Cholesky + explicit inverse per k ----------------
__global__ __launch_bounds__(256) void chol_kernel(const float* __restrict__ mut,
                                                   const float* __restrict__ logdetp,
                                                   float* __restrict__ minvU,
                                                   float* __restrict__ c1) {
  const int k = blockIdx.x * 256 + threadIdx.x;
  if (k >= KK) return;
  float Mm[8][8], L[8][8], Li[8][8];
  {
    int idx = 0;
    #pragma unroll
    for (int i = 0; i < 8; ++i)
      #pragma unroll
      for (int j = 0; j < 8; ++j)
        if (j >= i) {
          float v = mut[(size_t)k * 36 + idx++];
          Mm[i][j] = v; Mm[j][i] = v;
        }
  }
  #pragma unroll
  for (int i = 0; i < 8; ++i) Mm[i][i] += 1.0f;
  #pragma unroll
  for (int i = 0; i < 8; ++i)
    #pragma unroll
    for (int j = 0; j < 8; ++j) { L[i][j] = 0.0f; Li[i][j] = 0.0f; }

  float ld = 0.0f;
  #pragma unroll
  for (int j = 0; j < 8; ++j) {
    float sd = Mm[j][j];
    #pragma unroll
    for (int p = 0; p < 8; ++p) if (p < j) sd -= L[j][p] * L[j][p];
    float dj = sqrtf(fmaxf(sd, 1e-30f));
    L[j][j] = dj;
    ld += logf(dj);
    #pragma unroll
    for (int i = 0; i < 8; ++i) if (i > j) {
      float sv = Mm[i][j];
      #pragma unroll
      for (int p = 0; p < 8; ++p) if (p < j) sv -= L[i][p] * L[j][p];
      L[i][j] = sv / dj;
    }
  }
  #pragma unroll
  for (int c = 0; c < 8; ++c) {
    Li[c][c] = 1.0f / L[c][c];
    #pragma unroll
    for (int i = 0; i < 8; ++i) if (i > c) {
      float sv = 0.0f;
      #pragma unroll
      for (int p = 0; p < 8; ++p) if (p >= c && p < i) sv += L[i][p] * Li[p][c];
      Li[i][c] = -sv / L[i][i];
    }
  }
  {
    int idx = 0;
    #pragma unroll
    for (int i = 0; i < 8; ++i)
      #pragma unroll
      for (int j = 0; j < 8; ++j)
        if (j >= i) {
          float sv = 0.0f;
          #pragma unroll
          for (int p = 0; p < 8; ++p) sv += Li[p][i] * Li[p][j];
          minvU[(size_t)k * 36 + idx++] = sv;
        }
  }
  c1[k] = -0.5f * ((float)DD * LOG2PI + logdetp[0] + 2.0f * ld);
}

// ---------------- kernel 5: log_softmax(pi) folded into ck ----------------
__global__ __launch_bounds__(512) void ck_kernel(const float* __restrict__ pi_logits,
                                                 const float* __restrict__ c1,
                                                 float* __restrict__ ck) {
  __shared__ float red[512];
  const int t = threadIdx.x;
  float p = pi_logits[t];
  red[t] = p; __syncthreads();
  #pragma unroll
  for (int off = 256; off > 0; off >>= 1) {
    if (t < off) red[t] = fmaxf(red[t], red[t + off]);
    __syncthreads();
  }
  float mx = red[0]; __syncthreads();
  red[t] = __expf(p - mx); __syncthreads();
  #pragma unroll
  for (int off = 256; off > 0; off >>= 1) {
    if (t < off) red[t] += red[t + off];
    __syncthreads();
  }
  float logZ = mx + logf(red[0]);
  ck[t] = c1[t] + p - logZ;
}

// ---------------- kernel 6: main fused WMMA GEMM + epilogue ----------------
// grid = (64, KSPLIT); block = 256 (8 waves, one 16-row tile each).
// Each block covers 64 components (4 chunks of 16); double-buffered async LDS staging.
__global__ __launch_bounds__(256) void mfa_gemm_kernel(const unsigned short* __restrict__ xp,
                                                       const unsigned short* __restrict__ gwp,
                                                       const unsigned short* __restrict__ gmp,
                                                       const float* __restrict__ xq,
                                                       const float* __restrict__ wtpm,
                                                       const float* __restrict__ mupm,
                                                       const float* __restrict__ minvU,
                                                       const float* __restrict__ ck,
                                                       float* __restrict__ partMS) {
  __shared__ __align__(32) unsigned short stage[2][4608];
  __shared__ float epiV[8][256];
  __shared__ float epiU[8][256];

  const int t = threadIdx.x;
  const int w = t >> 5, lane = t & 31;
  const int rowTile = blockIdx.x * 8 + w;                     // 0..511
  const int ks = blockIdx.y;                                  // 0..7
  const unsigned short* aBase = xp + (size_t)rowTile * 64 * 512;

  float m = -INFINITY, s = 0.0f;
  const int row = lane >> 1, sel = lane & 1;
  const int rbase = (lane >> 4) * 8, colc = lane & 15;

  #pragma unroll 1
  for (int cc = 0; cc < 4; ++cc) {
    const int c = ks * 4 + cc;                                // chunk of 16 components
    v8f accW[8];
    v8f accU = {0, 0, 0, 0, 0, 0, 0, 0};
    #pragma unroll
    for (int j = 0; j < 8; ++j) accW[j] = (v8f){0, 0, 0, 0, 0, 0, 0, 0};

    // prologue: async-fill buffer 0 with dblk 0
    async_stage(&stage[0][0],
                gwp + ((size_t)0 * 256 + c * 8) * 512,
                gmp + ((size_t)0 * 32 + c) * 512, t);
    int cur = 0;

    #pragma unroll 1
    for (int dblk = 0; dblk < 64; ++dblk) {
      asm volatile("s_wait_asynccnt 0x0" ::: "memory");
      __syncthreads();                                        // stage[cur] ready; other buf free
      if (dblk < 63)
        async_stage(&stage[cur ^ 1][0],
                    gwp + ((size_t)(dblk + 1) * 256 + c * 8) * 512,
                    gmp + ((size_t)(dblk + 1) * 32 + c) * 512, t);

      v16bf af = *reinterpret_cast<const v16bf*>(aBase + (size_t)dblk * 512 + lane * 16);
      const unsigned short* sb = &stage[cur][0];
      v16bf bfr[9];
      #pragma unroll
      for (int j = 0; j < 8; ++j)
        bfr[j] = *reinterpret_cast<const v16bf*>(sb + j * 512 + lane * 16);
      bfr[8] = *reinterpret_cast<const v16bf*>(sb + 4096 + lane * 16);

      #pragma unroll
      for (int j = 0; j < 8; ++j)
        accW[j] = __builtin_amdgcn_wmma_f32_16x16x32_bf16(false, af, false, bfr[j],
                                                          (short)0, accW[j], false, false);
      accU = __builtin_amdgcn_wmma_f32_16x16x32_bf16(false, af, false, bfr[8],
                                                     (short)0, accU, false, false);
      cur ^= 1;
    }

    // ---- epilogue: v^T Minv v + quad_Psi + online logsumexp ----
    #pragma unroll
    for (int r = 0; r < 8; ++r) epiU[w][(rbase + r) * 16 + colc] = accU[r];

    #pragma unroll 1
    for (int j = 0; j < 8; ++j) {
      __syncthreads();
      #pragma unroll
      for (int r = 0; r < 8; ++r) epiV[w][(rbase + r) * 16 + colc] = accW[j][r];
      __syncthreads();

      const int k = c * 16 + j * 2 + sel;
      float v[8];
      #pragma unroll
      for (int q = 0; q < 8; ++q)
        v[q] = epiV[w][row * 16 + sel * 8 + q] - wtpm[(size_t)k * 8 + q];
      float u = epiU[w][row * 16 + j * 2 + sel];
      const int b = rowTile * 16 + row;
      float quadPsi = xq[b] - 2.0f * u + mupm[k];
      const float* Mi = minvU + (size_t)k * 36;
      float corr = 0.0f;
      int idx = 0;
      #pragma unroll
      for (int i = 0; i < 8; ++i) {
        corr += Mi[idx++] * v[i] * v[i];
        #pragma unroll
        for (int jj = i + 1; jj < 8; ++jj) corr += 2.0f * Mi[idx++] * v[i] * v[jj];
      }
      float ll = ck[k] - 0.5f * (quadPsi - corr);
      if (ll > m) { s = s * __expf(m - ll) + 1.0f; m = ll; }
      else        { s += __expf(ll - m); }
    }
  }

  // combine the two lanes per row, write partial (max, sum) for this K-split
  __syncthreads();
  epiU[w][lane * 2] = m;
  epiU[w][lane * 2 + 1] = s;
  __syncthreads();
  if (lane < 16) {
    float m0 = epiU[w][lane * 4 + 0], s0 = epiU[w][lane * 4 + 1];
    float m1 = epiU[w][lane * 4 + 2], s1 = epiU[w][lane * 4 + 3];
    float M = fmaxf(m0, m1);
    float S = s0 * __expf(m0 - M) + s1 * __expf(m1 - M);
    size_t idx = ((size_t)ks * BB + (size_t)rowTile * 16 + lane) * 2;
    partMS[idx] = M;
    partMS[idx + 1] = S;
  }
}

// ---------------- kernel 7: merge K-split partials, mean(-logp) ----------------
__global__ __launch_bounds__(256) void final_kernel(const float* __restrict__ partMS,
                                                    float* __restrict__ out) {
  __shared__ float red[256];
  const int t = threadIdx.x;
  float acc = 0.0f;
  for (int b = t; b < BB; b += 256) {
    float M = -INFINITY, S = 0.0f;
    #pragma unroll
    for (int ks = 0; ks < KSPLIT; ++ks) {
      float mm = partMS[((size_t)ks * BB + b) * 2];
      float ss = partMS[((size_t)ks * BB + b) * 2 + 1];
      if (mm > M) { S = S * __expf(M - mm) + ss; M = mm; }
      else        { S += ss * __expf(mm - M); }
    }
    acc += M + logf(S);
  }
  float s = block_reduce256(acc, red);
  if (t == 0) out[0] = -s / (float)BB;
}

// ---------------- launch ----------------
extern "C" void kernel_launch(void* const* d_in, const int* in_sizes, int n_in,
                              void* d_out, int out_size, void* d_ws, size_t ws_size,
                              hipStream_t stream) {
  const float* x         = (const float*)d_in[0];
  const float* mu        = (const float*)d_in[1];
  const float* dir_raw   = (const float*)d_in[2];
  const float* scale_rho = (const float*)d_in[3];
  const float* psi_rho   = (const float*)d_in[4];
  const float* pi_logits = (const float*)d_in[5];
  float* out = (float*)d_out;
  char* ws = (char*)d_ws;

  constexpr size_t OFF_PSIINV = 0;                                    // 2048 f
  constexpr size_t OFF_LOGDP  = OFF_PSIINV + 2048 * 4;                // 1 f
  constexpr size_t OFF_XQ     = OFF_LOGDP + 256;                      // 8192 f
  constexpr size_t OFF_MUPM   = OFF_XQ + (size_t)BB * 4;              // 512 f
  constexpr size_t OFF_WTPM   = OFF_MUPM + KK * 4;                    // 4096 f
  constexpr size_t OFF_MUT    = OFF_WTPM + (size_t)KK * 8 * 4;        // K*36 f
  constexpr size_t OFF_MINV   = OFF_MUT + (size_t)KK * 36 * 4;        // K*36 f
  constexpr size_t OFF_C1     = OFF_MINV + (size_t)KK * 36 * 4;       // 512 f
  constexpr size_t OFF_CK     = OFF_C1 + KK * 4;                      // 512 f
  constexpr size_t OFF_PART   = OFF_CK + KK * 4;                      // KSPLIT*B*2 f
  constexpr size_t OFF_XPACK  = OFF_PART + (size_t)KSPLIT * BB * 2 * 4;
  constexpr size_t OFF_GWP    = OFF_XPACK + (size_t)(BB / 16) * 64 * 512 * 2;
  constexpr size_t OFF_GMUP   = OFF_GWP + (size_t)64 * 256 * 512 * 2;

  float* psi_inv = (float*)(ws + OFF_PSIINV);
  float* logdetp = (float*)(ws + OFF_LOGDP);
  float* xq      = (float*)(ws + OFF_XQ);
  float* mupm    = (float*)(ws + OFF_MUPM);
  float* wtpm    = (float*)(ws + OFF_WTPM);
  float* mut     = (float*)(ws + OFF_MUT);
  float* minvU   = (float*)(ws + OFF_MINV);
  float* c1      = (float*)(ws + OFF_C1);
  float* ckv     = (float*)(ws + OFF_CK);
  float* part    = (float*)(ws + OFF_PART);
  unsigned short* xp  = (unsigned short*)(ws + OFF_XPACK);
  unsigned short* gwp = (unsigned short*)(ws + OFF_GWP);
  unsigned short* gmp = (unsigned short*)(ws + OFF_GMUP);

  psi_kernel<<<1, 256, 0, stream>>>(psi_rho, psi_inv, logdetp);
  pack_x_kernel<<<BB, 256, 0, stream>>>(x, psi_inv, xp, xq);
  prep_w_kernel<<<KK, 256, 0, stream>>>(mu, dir_raw, scale_rho, psi_inv,
                                        gwp, gmp, mut, wtpm, mupm);
  chol_kernel<<<2, 256, 0, stream>>>(mut, logdetp, minvU, c1);
  ck_kernel<<<1, 512, 0, stream>>>(pi_logits, c1, ckv);
  mfa_gemm_kernel<<<dim3(64, KSPLIT), 256, 0, stream>>>(xp, gwp, gmp, xq, wtpm, mupm,
                                                        minvU, ckv, part);
  final_kernel<<<1, 256, 0, stream>>>(part, out);
}
